// hyena1d_35218731828037
// MI455X (gfx1250) — matrix operand order repaired and checked by reading, not compiled, not run on hardware
//
#include <hip/hip_runtime.h>
#include <hip/hip_bf16.h>
#include <math.h>

// ---------------- CDNA5 WMMA / TDM types ----------------
typedef __attribute__((ext_vector_type(16))) _Float16     v16h;
typedef __attribute__((ext_vector_type(8)))  float        v8f;
typedef __attribute__((ext_vector_type(4)))  unsigned int v4u;
typedef __attribute__((ext_vector_type(8)))  int          v8i;
typedef __attribute__((ext_vector_type(4)))  int          v4i;

#define BB 8
#define LL 4096
#define DD 128
#define TD 384           // 3*D
#define FOW 64           // filter MLP width
#define NLAYERS 8
#define BLROWS (BB*LL)   // 32768 rows of the (B,L,*) matrices

// =====================================================================
// TDM: 1-D tile load (Global -> LDS), elements are 2 bytes (f16).
// Descriptor built per CDNA5 ISA 8.3-8.6 (count=1, type=2, data_size=1(2B),
// tensor_dim0=tile_dim0=n, stride0=n). Wave-level op, tracked by TENSORcnt.
// =====================================================================
__device__ __forceinline__ void hy_tdm_load_2b(unsigned lds_addr, const void* gptr,
                                               unsigned n /* #f16 elems, <=65535 */) {
    unsigned long long ga = (unsigned long long)gptr;
    v4u g0;
    g0[0] = 1u;                                        // count=1, user descriptor
    g0[1] = lds_addr;                                  // LDS byte address
    g0[2] = (unsigned)(ga & 0xffffffffu);              // global_addr[31:0]
    g0[3] = (unsigned)((ga >> 32) & 0x01ffffffu) | (2u << 30);  // [56:32] | type=2
    v8i g1;
    g1[0] = (int)(1u << 16);                           // data_size=1 (2 bytes)
    g1[1] = (int)((n & 0xffffu) << 16);                // tensor_dim0[15:0]
    g1[2] = (int)(((n >> 16) & 0xffffu) | (1u << 16)); // tensor_dim0[31:16] | tensor_dim1=1
    g1[3] = (int)((n & 0xffffu) << 16);                // tile_dim0 = n
    g1[4] = 0;                                         // tile_dim1/2 unused
    g1[5] = (int)n;                                    // tensor_dim0_stride[31:0]
    g1[6] = 0;
    g1[7] = 0;
    v4i z = {0, 0, 0, 0};
#if __has_include(<hip/amd_detail/amd_gfx1250_TDM.h>)
    v8i z8 = {0, 0, 0, 0, 0, 0, 0, 0};                 // amdgpu-toolchain 6-arg form
    __builtin_amdgcn_tensor_load_to_lds(g0, g1, z, z, z8, 0);
#else
    __builtin_amdgcn_tensor_load_to_lds(g0, g1, z, z, 0);   // ROCm 7.2 5-arg form
#endif
}

// =====================================================================
// Utility kernels
// =====================================================================
__global__ void hy_zero(float* p, long n) {
    long i = (long)blockIdx.x * blockDim.x + threadIdx.x;
    if (i < n) p[i] = 0.f;
}

__global__ void hy_cvt_f16(const float* s, _Float16* d, long n) {
    long i = (long)blockIdx.x * blockDim.x + threadIdx.x;
    if (i < n) d[i] = (_Float16)s[i];
}

// InstanceNorm over last dim (D=128), one wave32 per row; optional residual add.
__global__ void hy_inorm(const float* src, const float* resid, _Float16* dst, int rows) {
    int wid  = (int)(((long)blockIdx.x * blockDim.x + threadIdx.x) >> 5);
    int lane = threadIdx.x & 31;
    if (wid >= rows) return;
    const float* r = src + (long)wid * DD;
    float v[4]; float s = 0.f;
#pragma unroll
    for (int i = 0; i < 4; ++i) { v[i] = r[lane + 32 * i]; s += v[i]; }
#pragma unroll
    for (int m = 16; m; m >>= 1) s += __shfl_xor(s, m, 32);
    float mean = s * (1.f / 128.f);
    float s2 = 0.f;
#pragma unroll
    for (int i = 0; i < 4; ++i) { float d = v[i] - mean; s2 += d * d; }
#pragma unroll
    for (int m = 16; m; m >>= 1) s2 += __shfl_xor(s2, m, 32);
    float rstd = rsqrtf(s2 * (1.f / 128.f) + 1e-5f);
    _Float16* o = dst + (long)wid * DD;
    const float* xr = resid ? (resid + (long)wid * DD) : nullptr;
#pragma unroll
    for (int i = 0; i < 4; ++i) {
        float val = (v[i] - mean) * rstd;
        if (xr) val += xr[lane + 32 * i];
        o[lane + 32 * i] = (_Float16)val;
    }
}

// =====================================================================
// Implicit filter MLP: z(L,3) -> 64 -> 64 -> 64 -> 128, sin activations,
// exponential modulation; output kf (D, L) f16 filter taps.
// =====================================================================
__global__ void hy_filter(const float* W0, const float* b0, const float* fr0,
                          const float* W1, const float* b1, const float* fr1,
                          const float* W2, const float* b2, const float* fr2,
                          const float* Wout, _Float16* kf) {
    int l = blockIdx.x * blockDim.x + threadIdx.x;
    if (l >= LL) return;
    const float PI2 = 6.283185307179586f;
    float tl  = (float)l / (float)(LL - 1);
    float wph = 1e-4f * PI2 * (float)l / (float)LL;
    float z0 = tl, z1 = cosf(wph), z2 = -sinf(wph);

    float ha[FOW], hb[FOW];
    for (int j = 0; j < FOW; ++j)
        ha[j] = sinf(fr0[j] * (W0[j * 3 + 0] * z0 + W0[j * 3 + 1] * z1 +
                               W0[j * 3 + 2] * z2 + b0[j]));
    for (int j = 0; j < FOW; ++j) {
        float acc = b1[j];
        for (int i = 0; i < FOW; ++i) acc += W1[j * FOW + i] * ha[i];
        hb[j] = sinf(fr1[j] * acc);
    }
    for (int j = 0; j < FOW; ++j) {
        float acc = b2[j];
        for (int i = 0; i < FOW; ++i) acc += W2[j * FOW + i] * hb[i];
        ha[j] = sinf(fr2[j] * acc);
    }
    const float dmin = -3.0701134573253944f;   // ln(1e-2)/1.5
    const float dmax = -15.350567286626972f;   // ln(1e-2)/0.3
    for (int c = 0; c < DD; ++c) {
        float acc = 0.f;
        for (int i = 0; i < FOW; ++i) acc += Wout[c * FOW + i] * ha[i];
        float delta = fabsf(dmin + (dmax - dmin) * ((float)c / 127.f));
        kf[(long)c * LL + l] = (_Float16)(acc * expf(-tl * delta));
    }
}

// =====================================================================
// WMMA GEMM: C(MxN) = A(MxK,f16 rm) * W(NxK,f16 rm)^T + bias
// One wave computes a 16x64 C strip (1 A fragment feeds 4 WMMAs).
// flags: 1 = exact GELU epilogue, 2 = accumulate into Cf.  N % 64 == 0.
// =====================================================================
__global__ void hy_gemm(const _Float16* __restrict__ A, const _Float16* __restrict__ W,
                        const float* __restrict__ bias, float* Cf, _Float16* Ch,
                        int M, int N, int K, int flags) {
    int wave = threadIdx.x >> 5;
    int lane = threadIdx.x & 31;
    int ngrp = N >> 6;                          // groups of 4 N-tiles
    long tile = (long)blockIdx.x * (blockDim.x >> 5) + wave;
    long total = (long)(M >> 4) * ngrp;
    if (tile >= total) return;                  // wave-uniform: EXEC stays full
    int mt = (int)(tile / ngrp), ng = (int)(tile % ngrp);
    int khalf = lane >> 4;
    int arow  = mt * 16 + (lane & 15);
    int ncol0 = ng * 64 + (lane & 15);

    v8f acc0 = {}, acc1 = {}, acc2 = {}, acc3 = {};
    for (int k0 = 0; k0 < K; k0 += 32) {
        v16h a, b0, b1, b2, b3;
        const _Float16* Ar = A + (long)arow * K + k0;
#pragma unroll
        for (int j = 0; j < 16; ++j) {          // A 16x32 f16: contiguous k = 16*khalf + j
            int kk = j + 16 * khalf;
            a[j] = Ar[kk];
        }
        const _Float16* Wc = W + (long)ncol0 * K + k0 + 16 * khalf;
        __builtin_prefetch(Wc + 32);            // gfx1250 global_prefetch_b8
#pragma unroll
        for (int j = 0; j < 16; ++j) b0[j] = Wc[j];
#pragma unroll
        for (int j = 0; j < 16; ++j) b1[j] = Wc[16 * K + j];
#pragma unroll
        for (int j = 0; j < 16; ++j) b2[j] = Wc[32 * K + j];
#pragma unroll
        for (int j = 0; j < 16; ++j) b3[j] = Wc[48 * K + j];
        acc0 = __builtin_amdgcn_wmma_f32_16x16x32_f16(false, a, false, b0, (short)0, acc0, false, false);
        acc1 = __builtin_amdgcn_wmma_f32_16x16x32_f16(false, a, false, b1, (short)0, acc1, false, false);
        acc2 = __builtin_amdgcn_wmma_f32_16x16x32_f16(false, a, false, b2, (short)0, acc2, false, false);
        acc3 = __builtin_amdgcn_wmma_f32_16x16x32_f16(false, a, false, b3, (short)0, acc3, false, false);
    }
#pragma unroll
    for (int t = 0; t < 4; ++t) {
        v8f acc = (t == 0) ? acc0 : (t == 1) ? acc1 : (t == 2) ? acc2 : acc3;
        int ncol = ncol0 + 16 * t;
        float bv = bias ? bias[ncol] : 0.f;
#pragma unroll
        for (int r = 0; r < 8; ++r) {           // C: VGPR r <-> row r (+8 high half)
            int orow = mt * 16 + r + 8 * khalf;
            float v = acc[r] + bv;
            if (flags & 1) v = 0.5f * v * (1.f + erff(v * 0.7071067811865476f));
            long idx = (long)orow * N + ncol;
            if (Cf) { if (flags & 2) Cf[idx] += v; else Cf[idx] = v; }
            if (Ch) Ch[idx] = (_Float16)v;
        }
    }
}

// =====================================================================
// Depthwise causal conv (k=3) over L on u(B,L,3D) + split + gate:
//   x0 -> (B,L,D) f32 ; vgT = (v*x1) -> (D,L,8) f16
// =====================================================================
__global__ void hy_sfilter(const float* __restrict__ u, const float* __restrict__ sfW,
                           const float* __restrict__ sfb, float* x0, _Float16* vgT) {
    long idx = (long)blockIdx.x * blockDim.x + threadIdx.x;   // over B*L*D
    if (idx >= (long)BLROWS * DD) return;
    int d = (int)(idx % DD);
    long bl = idx / DD;
    int l = (int)(bl % LL);
    int b = (int)(bl / LL);
    const float* ub = u + ((long)b * LL + l) * TD;
    auto conv3 = [&](int c) -> float {
        float acc = sfb[c] + sfW[c * 3 + 2] * ub[c];
        if (l >= 1) acc += sfW[c * 3 + 1] * ub[c - TD];
        if (l >= 2) acc += sfW[c * 3 + 0] * ub[c - 2 * TD];
        return acc;
    };
    float a0 = conv3(d);
    float a1 = conv3(DD + d);
    float av = conv3(2 * DD + d);
    x0[idx] = a0;
    vgT[((long)d * LL + l) * 8 + b] = (_Float16)(av * a1);
}

// =====================================================================
// Long causal conv as lower-triangular Toeplitz block-GEMM on the WMMA pipe.
// Grid: (32 mt-groups, 128 channels); one workgroup owns channel d.
// Wave 0 stages the filter prefix kd[0..nl) and input panel ud[0..nl*8) into
// LDS with two TDM tensor_load_to_lds DMAs, waits TENSORcnt, block barriers;
// 8 waves then each build one 16-row output tile from LDS fragments.
// Dynamic LDS: 8KB taps + 64KB panel = 73728 B.
// =====================================================================
__global__ void hy_conv(const _Float16* __restrict__ kf, const _Float16* __restrict__ vgT,
                        float* __restrict__ y) {
    extern __shared__ _Float16 smem[];          // [0,4096) taps, [4096,36864) panel
    int wave = threadIdx.x >> 5;
    int lane = threadIdx.x & 31;
    int bx = blockIdx.x;                        // mt group (8 tiles per block)
    int d  = blockIdx.y;                        // channel
    int nl = (bx + 1) * 128;                    // staged prefix length (<= LL)

    unsigned lds_base = (unsigned)(size_t)(void*)smem;   // LDS byte offset (addr[31:0])
    if (wave == 0) {
        hy_tdm_load_2b(lds_base,        kf  + (long)d * LL,     (unsigned)nl);
        hy_tdm_load_2b(lds_base + 8192, vgT + (long)d * LL * 8, (unsigned)(nl * 8));
        __builtin_amdgcn_s_wait_tensorcnt(0);
    }
    __syncthreads();

    const _Float16* kd = smem;                  // taps
    const _Float16* ud = smem + 4096;           // (s, batch) panel

    int mt    = bx * 8 + wave;                  // this wave's output row tile
    int khalf = lane >> 4;
    int trow  = mt * 16 + (lane & 15);          // output time (A row)
    int ncol  = lane & 15;                      // batch column (8 valid)

    v8f acc = {};
    int kend = mt * 16 + 16;                    // causal: s <= t_max
    for (int k0 = 0; k0 < kend; k0 += 32) {
        v16h a, b;
#pragma unroll
        for (int j = 0; j < 16; ++j) {
            int kk = j + 16 * khalf;
            int diff = trow - (k0 + kk);
            a[j] = (diff >= 0) ? kd[diff] : (_Float16)0.f;
        }
        int sbase = k0 + 16 * khalf;
#pragma unroll
        for (int j = 0; j < 16; ++j) {
            int s = sbase + j;                  // mask s>=kend: pairs with zero A entries
            b[j] = (ncol < 8 && s < kend) ? ud[s * 8 + ncol] : (_Float16)0.f;
        }
        acc = __builtin_amdgcn_wmma_f32_16x16x32_f16(false, a, false, b,
                                                     (short)0, acc, false, false);
    }
    if (ncol < 8) {
#pragma unroll
        for (int r = 0; r < 8; ++r) {
            int t = mt * 16 + r + 8 * khalf;
            y[((long)d * LL + t) * 8 + ncol] = acc[r];
        }
    }
}

// fftconv skip-bias + pre-out-proj gate:  g = (conv + vg*bias[d]) * x0
__global__ void hy_postgate(const float* __restrict__ y, const _Float16* __restrict__ vgT,
                            const float* __restrict__ fbias, const float* __restrict__ x0,
                            _Float16* g16) {
    long idx = (long)blockIdx.x * blockDim.x + threadIdx.x;   // over B*L*D
    if (idx >= (long)BLROWS * DD) return;
    int d = (int)(idx % DD);
    long bl = idx / DD;
    int l = (int)(bl % LL);
    int b = (int)(bl / LL);
    long t = ((long)d * LL + l) * 8 + b;
    float vc = y[t] + (float)vgT[t] * fbias[d];
    g16[idx] = (_Float16)(vc * x0[idx]);
}

// =====================================================================
// Host orchestration
// =====================================================================
extern "C" void kernel_launch(void* const* d_in, const int* in_sizes, int n_in,
                              void* d_out, int out_size, void* d_ws, size_t ws_size,
                              hipStream_t stream) {
    (void)in_sizes; (void)n_in; (void)out_size; (void)ws_size;
    const float* x      = (const float*)d_in[0];
    const float* in_W   = (const float*)d_in[1];   // (8,384,128)
    const float* in_b   = (const float*)d_in[2];   // (8,384)
    const float* sf_W   = (const float*)d_in[3];   // (8,384,3)
    const float* sf_b   = (const float*)d_in[4];   // (8,384)
    const float* W0     = (const float*)d_in[5];   // (8,64,3)
    const float* b0     = (const float*)d_in[6];
    const float* fr0    = (const float*)d_in[7];
    const float* W1     = (const float*)d_in[8];   // (8,64,64)
    const float* b1     = (const float*)d_in[9];
    const float* fr1    = (const float*)d_in[10];
    const float* W2     = (const float*)d_in[11];  // (8,64,64)
    const float* b2     = (const float*)d_in[12];
    const float* fr2    = (const float*)d_in[13];
    const float* Wout   = (const float*)d_in[14];  // (8,128,64)
    const float* fltb   = (const float*)d_in[15];  // (8,128)
    const float* out_W  = (const float*)d_in[16];  // (8,128,128)
    const float* out_b  = (const float*)d_in[17];  // (8,128)
    const float* f_W1   = (const float*)d_in[18];  // (8,256,128)
    const float* f_b1   = (const float*)d_in[19];  // (8,256)
    const float* f_W2   = (const float*)d_in[20];  // (8,128,256)
    const float* f_b2   = (const float*)d_in[21];  // (8,128)
    float* out = (float*)d_out;

    size_t off = 0;
    auto wsa = [&](size_t bytes) -> void* {
        void* p = (char*)d_ws + off;
        off += (bytes + 255) & ~(size_t)255;
        return p;
    };
    _Float16* xn16  = (_Float16*)wsa((size_t)BLROWS * DD * 2);
    float*    u     = (float*)   wsa((size_t)BLROWS * TD * 4);
    float*    x0    = (float*)   wsa((size_t)BLROWS * DD * 4);
    _Float16* vgT   = (_Float16*)wsa((size_t)DD * LL * 8 * 2);
    _Float16* kf    = (_Float16*)wsa((size_t)DD * LL * 2);
    float*    yc    = (float*)   wsa((size_t)DD * LL * 8 * 4);
    _Float16* g16   = (_Float16*)wsa((size_t)BLROWS * DD * 2);
    float*    y2    = (float*)   wsa((size_t)BLROWS * DD * 4);
    _Float16* h16   = (_Float16*)wsa((size_t)BLROWS * DD * 2);
    _Float16* a2h   = (_Float16*)wsa((size_t)BLROWS * 2 * DD * 2);
    _Float16* inWh  = (_Float16*)wsa((size_t)TD * DD * 2);
    _Float16* outWh = (_Float16*)wsa((size_t)DD * DD * 2);
    _Float16* fW1h  = (_Float16*)wsa((size_t)2 * DD * DD * 2);
    _Float16* fW2h  = (_Float16*)wsa((size_t)DD * 2 * DD * 2);

    const long nBLD = (long)BLROWS * DD;
    hy_zero<<<(int)((nBLD + 255) / 256), 256, 0, stream>>>(out, nBLD);
    hy_inorm<<<(BLROWS + 7) / 8, 256, 0, stream>>>(x, nullptr, xn16, BLROWS);

    for (int li = 0; li < NLAYERS; ++li) {
        const float* inW_l  = in_W  + (size_t)li * TD * DD;
        const float* inb_l  = in_b  + (size_t)li * TD;
        const float* sfW_l  = sf_W  + (size_t)li * TD * 3;
        const float* sfb_l  = sf_b  + (size_t)li * TD;
        const float* W0_l   = W0    + (size_t)li * FOW * 3;
        const float* b0_l   = b0    + (size_t)li * FOW;
        const float* fr0_l  = fr0   + (size_t)li * FOW;
        const float* W1_l   = W1    + (size_t)li * FOW * FOW;
        const float* b1_l   = b1    + (size_t)li * FOW;
        const float* fr1_l  = fr1   + (size_t)li * FOW;
        const float* W2_l   = W2    + (size_t)li * FOW * FOW;
        const float* b2_l   = b2    + (size_t)li * FOW;
        const float* fr2_l  = fr2   + (size_t)li * FOW;
        const float* Wout_l = Wout  + (size_t)li * DD * FOW;
        const float* fb_l   = fltb  + (size_t)li * DD;
        const float* oW_l   = out_W + (size_t)li * DD * DD;
        const float* ob_l   = out_b + (size_t)li * DD;
        const float* fW1_l  = f_W1  + (size_t)li * 2 * DD * DD;
        const float* fb1_l  = f_b1  + (size_t)li * 2 * DD;
        const float* fW2_l  = f_W2  + (size_t)li * DD * 2 * DD;
        const float* fb2_l  = f_b2  + (size_t)li * DD;

        hy_cvt_f16<<<(TD * DD + 255) / 256, 256, 0, stream>>>(inW_l, inWh, TD * DD);
        hy_cvt_f16<<<(DD * DD + 255) / 256, 256, 0, stream>>>(oW_l, outWh, DD * DD);
        hy_cvt_f16<<<(2 * DD * DD + 255) / 256, 256, 0, stream>>>(fW1_l, fW1h, 2 * DD * DD);
        hy_cvt_f16<<<(2 * DD * DD + 255) / 256, 256, 0, stream>>>(fW2_l, fW2h, 2 * DD * DD);

        hy_filter<<<LL / 128, 128, 0, stream>>>(W0_l, b0_l, fr0_l, W1_l, b1_l, fr1_l,
                                                W2_l, b2_l, fr2_l, Wout_l, kf);

        long tiles = (long)(BLROWS / 16) * (TD / 64);
        hy_gemm<<<(int)((tiles + 7) / 8), 256, 0, stream>>>(xn16, inWh, inb_l, u, nullptr,
                                                            BLROWS, TD, DD, 0);

        hy_sfilter<<<(int)((nBLD + 255) / 256), 256, 0, stream>>>(u, sfW_l, sfb_l, x0, vgT);

        hy_conv<<<dim3(LL / 16 / 8, DD), 256, 73728, stream>>>(kf, vgT, yc);

        hy_postgate<<<(int)((nBLD + 255) / 256), 256, 0, stream>>>(yc, vgT, fb_l, x0, g16);

        tiles = (long)(BLROWS / 16) * (DD / 64);
        hy_gemm<<<(int)((tiles + 7) / 8), 256, 0, stream>>>(g16, outWh, ob_l, y2, nullptr,
                                                            BLROWS, DD, DD, 0);

        hy_inorm<<<(BLROWS + 7) / 8, 256, 0, stream>>>(y2, x, h16, BLROWS);

        tiles = (long)(BLROWS / 16) * ((2 * DD) / 64);
        hy_gemm<<<(int)((tiles + 7) / 8), 256, 0, stream>>>(h16, fW1h, fb1_l, nullptr, a2h,
                                                            BLROWS, 2 * DD, DD, 1 /*gelu*/);

        tiles = (long)(BLROWS / 16) * (DD / 64);
        hy_gemm<<<(int)((tiles + 7) / 8), 256, 0, stream>>>(a2h, fW2h, fb2_l, out, nullptr,
                                                            BLROWS, DD, 2 * DD, 2 /*accum*/);
    }
}
